// TransformerBlock_67894843015261
// MI455X (gfx1250) — matrix operand order repaired and checked by reading, not compiled, not run on hardware
//
#include <hip/hip_runtime.h>
#include <math.h>

// ---------------------------------------------------------------------------
// Transformer block for MI455X (gfx1250, wave32, WMMA, TDM).
//  * All GEMM-shaped math on v_wmma_f32_16x16x32_bf16 (bf16 in, f32 acc).
//  * Weights pre-transposed to [N,K] bf16 once so the B operand is
//    K-contiguous per lane -> global_load_b128 feeding v_wmma (weight slabs
//    live in the 192MB L2).
//  * Activation (A) tiles staged into LDS by the Tensor Data Mover
//    (tensor_load_to_lds, double-buffered, s_wait_tensorcnt) so the WMMA
//    loop overlaps with the async DMA; D# pads LDS rows to 80B for
//    conflict-free ds_load_b128 fragment reads.
//  * Flash-style attention: one wave per (batch, head, 16-query tile);
//    softmax row-sums computed by an extra WMMA against an all-ones matrix
//    (replaces 32 ds_bpermute per tile), row-max via lane butterflies.
//  * global_prefetch_b8 (__builtin_prefetch) hides HBM latency on B streams.
// ---------------------------------------------------------------------------

typedef __bf16 bf16_t;
typedef __attribute__((ext_vector_type(16))) __bf16 v16bf;
typedef __attribute__((ext_vector_type(8)))  __bf16 v8bf;
typedef __attribute__((ext_vector_type(8)))  float  v8f;
typedef __attribute__((ext_vector_type(4)))  unsigned int u32x4;
typedef __attribute__((ext_vector_type(8)))  unsigned int u32x8;

#define DIM_   1024
#define HEADS_ 16
#define DHEAD_ 64
#define INNER_ 1024
#define FF_    4096
#define BATCH_ 2
#define SEQ_   2048
#define ROWS_  (BATCH_ * SEQ_)   // 4096

// ----------------------------- fragment builders ---------------------------
// A-frag 16x32 bf16 from a K-contiguous row pointer (row = lane&15):
// lane<16: elems 0..15 = K0..15 ; lane>=16: K16..31.
__device__ inline v16bf make_a_frag(const bf16_t* __restrict__ row_k0, int kb8) {
  v8bf lo = *reinterpret_cast<const v8bf*>(row_k0 + 2 * kb8);        // 16B aligned
  v8bf hi = *reinterpret_cast<const v8bf*>(row_k0 + 2 * kb8 + 8);    // 16B aligned
  return __builtin_shufflevector(lo, hi, 0, 1, 2, 3, 4, 5, 6, 7,
                                 8, 9, 10, 11, 12, 13, 14, 15);
}
// B-frag 32x16 bf16 from a K-contiguous column pointer (col = lane&15):
// lanes 0..15 hold K=0..15, lanes 16..31 hold K=16..31 (16 contiguous elems).
__device__ inline v16bf make_b_frag(const bf16_t* __restrict__ col_k0, int kb16) {
  return *reinterpret_cast<const v16bf*>(col_k0 + kb16);         // 32B aligned
}

// 32-bit LDS byte offset of a __shared__ object (ptrtoint of AS(3) pointer).
__device__ inline unsigned lds_offset_of(const void* p) {
  return (unsigned)(unsigned long long)
      (const __attribute__((address_space(3))) void*)p;
}

// --------------------------- TDM tile load (D#) ----------------------------
// 2D tile: 32 bf16 per row (64B), TILE_ROWS rows, global row stride = K elems.
// pad_enable: +4 DWORDs (16B) after every 16 DWORDs (64B) -> 80B LDS rows.
__device__ inline void tdm_load_a_tile(unsigned lds_byte_addr,
                                       const bf16_t* gaddr, unsigned K_elems,
                                       unsigned tile_rows) {
  unsigned long long ga = (unsigned long long)gaddr;
  u32x4 g0;
  g0[0] = 1u;                                            // count=1, user mode
  g0[1] = lds_byte_addr;                                 // lds_addr
  g0[2] = (unsigned)(ga & 0xffffffffu);                  // global_addr[31:0]
  g0[3] = (unsigned)((ga >> 32) & 0x1ffffffu)            // global_addr[56:32]
        | (2u << 30);                                    // type = 2 ("image")
  u32x8 g1;
  const unsigned td0 = 32u;                              // tensor_dim0 (elems)
  const unsigned td1 = tile_rows;                        // tensor_dim1
  g1[0] = (1u << 16)                                     // data_size = 2B
        | (1u << 20)                                     // pad_enable
        | (3u << 22)                                     // pad_interval: 16 DW
        | (3u << 25);                                    // pad_amount: 4 DW
  g1[1] = (td0 & 0xffffu) << 16;                         // tensor_dim0 lo16
  g1[2] = (td0 >> 16) | ((td1 & 0xffffu) << 16);         // dim0 hi / dim1 lo
  g1[3] = (td1 >> 16) | (32u << 16);                     // dim1 hi, tile_dim0
  g1[4] = tile_rows;                                     // tile_dim1 (dim2=0)
  g1[5] = K_elems;                                       // dim0_stride[31:0]
  g1[6] = 0u;                                            // stride hi bits
  g1[7] = 0u;
  asm volatile("tensor_load_to_lds %0, %1" :: "s"(g0), "s"(g1) : "memory");
}

// ------------------------------- LayerNorm ---------------------------------
__global__ __launch_bounds__(256)
void ln_bf16_kernel(const float* __restrict__ x, const float* __restrict__ g,
                    const float* __restrict__ b, bf16_t* __restrict__ out) {
  const int dim = DIM_;
  long row = blockIdx.x;
  const float* xr = x + row * dim;
  float s = 0.f, s2 = 0.f;
  for (int i = threadIdx.x; i < dim; i += 256) {
    float v = xr[i];
    s += v; s2 += v * v;
  }
#pragma unroll
  for (int off = 16; off; off >>= 1) {
    s  += __shfl_down(s, off);
    s2 += __shfl_down(s2, off);
  }
  __shared__ float ps[8], ps2[8];
  int wave = threadIdx.x >> 5, lane = threadIdx.x & 31;
  if (lane == 0) { ps[wave] = s; ps2[wave] = s2; }
  __syncthreads();
  if (threadIdx.x == 0) {
    float t = 0.f, t2 = 0.f;
#pragma unroll
    for (int i = 0; i < 8; ++i) { t += ps[i]; t2 += ps2[i]; }
    ps[0] = t; ps2[0] = t2;
  }
  __syncthreads();
  float mean = ps[0] * (1.0f / dim);
  float var  = ps2[0] * (1.0f / dim) - mean * mean;
  float inv  = rsqrtf(var + 1e-5f);
  for (int i = threadIdx.x; i < dim; i += 256)
    out[row * dim + i] = (bf16_t)((xr[i] - mean) * inv * g[i] + b[i]);
}

// ------------------- f32 [R,C] -> bf16 [C,R] transpose ---------------------
__global__ __launch_bounds__(256)
void cvt_transpose_bf16_kernel(const float* __restrict__ in, bf16_t* __restrict__ out,
                               int R, int C) {
  __shared__ float tile[32][33];
  int c0 = blockIdx.x * 32;
  int r0 = blockIdx.y * 32;
  int tx = threadIdx.x & 31;
  int ty = threadIdx.x >> 5;   // 8 rows per pass
#pragma unroll
  for (int i = 0; i < 32; i += 8)
    tile[ty + i][tx] = in[(long)(r0 + ty + i) * C + c0 + tx];
  __syncthreads();
#pragma unroll
  for (int i = 0; i < 32; i += 8)
    out[(long)(c0 + ty + i) * R + r0 + tx] = (bf16_t)tile[tx][ty + i];
}

// ------------------------------- WMMA GEMM ---------------------------------
// C[M,N] = A[M,K](bf16, row-major) * Bt[N,K](bf16, pre-transposed weights)
// EP: 0 = plain, 1 = +bias, 2 = +bias then exact GELU, 3 = +bias +residual
// Block: 256 threads = 8 waves (4x2), block tile 128x128, wave tile 32x64.
// A tile (128x32) staged by the TDM into double-buffered LDS (80B rows).
template <int EP, typename OutT>
__global__ __launch_bounds__(256)
void gemm_bf16_kernel(const bf16_t* __restrict__ A, const bf16_t* __restrict__ Bt,
                      OutT* __restrict__ C, const float* __restrict__ bias,
                      const float* __restrict__ resid, int M, int N, int K) {
  __shared__ __align__(16) bf16_t Asmem[2][128][40];  // 80B rows, 2x10KB

  int wave = threadIdx.x >> 5;
  int wm   = wave & 3;   // 4 wave-rows of 32
  int wn   = wave >> 2;  // 2 wave-cols of 64
  int bm0  = blockIdx.y * 128 + wm * 32;
  int bn0  = blockIdx.x * 128 + wn * 64;

  int lane = threadIdx.x & 31;
  int l15  = lane & 15;
  int kb8  = (lane >> 4) * 8;
  int kb16 = (lane >> 4) * 16;

  const bf16_t* Ablk = A + (long)(blockIdx.y * 128) * K;  // block's A rows
  const bf16_t* brow0 = Bt + (long)(bn0 +  0 + l15) * K;
  const bf16_t* brow1 = Bt + (long)(bn0 + 16 + l15) * K;
  const bf16_t* brow2 = Bt + (long)(bn0 + 32 + l15) * K;
  const bf16_t* brow3 = Bt + (long)(bn0 + 48 + l15) * K;

  unsigned lds_buf[2] = { lds_offset_of(&Asmem[0][0][0]),
                          lds_offset_of(&Asmem[1][0][0]) };
  bool issuer = (threadIdx.x < 32);   // wave 0 drives the TDM

  if (issuer) tdm_load_a_tile(lds_buf[0], Ablk, (unsigned)K, 128u);

  v8f acc[2][4] = {};

  int buf = 0;
  for (int k0 = 0; k0 < K; k0 += 32, buf ^= 1) {
    if (issuer) {
      if (k0 + 32 < K) {
        tdm_load_a_tile(lds_buf[buf ^ 1], Ablk + (k0 + 32), (unsigned)K, 128u);
        __builtin_amdgcn_s_wait_tensorcnt(1);  // tile k0 has landed
      } else {
        __builtin_amdgcn_s_wait_tensorcnt(0);
      }
    }
    __syncthreads();   // LDS tile `buf` ready for everyone

    if (k0 + 64 < K) {  // prefetch next B K-slab -> global_prefetch_b8
      __builtin_prefetch(brow0 + k0 + 64, 0, 1);
      __builtin_prefetch(brow2 + k0 + 64, 0, 1);
    }

    v16bf a0 = make_a_frag(&Asmem[buf][wm * 32 + l15][0],      kb8);
    v16bf a1 = make_a_frag(&Asmem[buf][wm * 32 + 16 + l15][0], kb8);
    v16bf b0 = make_b_frag(brow0 + k0, kb16);
    v16bf b1 = make_b_frag(brow1 + k0, kb16);
    v16bf b2 = make_b_frag(brow2 + k0, kb16);
    v16bf b3 = make_b_frag(brow3 + k0, kb16);

    acc[0][0] = __builtin_amdgcn_wmma_f32_16x16x32_bf16(false, a0, false, b0, (short)0, acc[0][0], false, false);
    acc[0][1] = __builtin_amdgcn_wmma_f32_16x16x32_bf16(false, a0, false, b1, (short)0, acc[0][1], false, false);
    acc[0][2] = __builtin_amdgcn_wmma_f32_16x16x32_bf16(false, a0, false, b2, (short)0, acc[0][2], false, false);
    acc[0][3] = __builtin_amdgcn_wmma_f32_16x16x32_bf16(false, a0, false, b3, (short)0, acc[0][3], false, false);
    acc[1][0] = __builtin_amdgcn_wmma_f32_16x16x32_bf16(false, a1, false, b0, (short)0, acc[1][0], false, false);
    acc[1][1] = __builtin_amdgcn_wmma_f32_16x16x32_bf16(false, a1, false, b1, (short)0, acc[1][1], false, false);
    acc[1][2] = __builtin_amdgcn_wmma_f32_16x16x32_bf16(false, a1, false, b2, (short)0, acc[1][2], false, false);
    acc[1][3] = __builtin_amdgcn_wmma_f32_16x16x32_bf16(false, a1, false, b3, (short)0, acc[1][3], false, false);

    __syncthreads();   // everyone done with `buf` before TDM reuses it
  }

  int cn    = l15;
  int rbase = (lane >> 4) * 8;
#pragma unroll
  for (int i = 0; i < 2; ++i)
#pragma unroll
    for (int j = 0; j < 4; ++j) {
      int col = bn0 + j * 16 + cn;
      float bcol = (EP >= 1) ? bias[col] : 0.0f;
#pragma unroll
      for (int r = 0; r < 8; ++r) {
        long row = bm0 + i * 16 + rbase + r;
        float v = acc[i][j][r] + bcol;
        if (EP == 2) v = 0.5f * v * (1.0f + erff(v * 0.70710678118f));
        if (EP == 3) v += resid[row * N + col];
        C[row * N + col] = (OutT)v;
      }
    }
}

// ---------------------------- Flash attention ------------------------------
// One wave per (batch, head, 16-query tile). qkv: [B*N, 3*INNER] bf16,
// out: [B*N, INNER] bf16.  scale = 1/sqrt(64) = 0.125.
__global__ __launch_bounds__(32)
void flash_attn_kernel(const bf16_t* __restrict__ qkv, bf16_t* __restrict__ out) {
  const int N = SEQ_, D = DHEAD_;
  const int LDQ = 3 * INNER_;
  int idx = blockIdx.x;
  int qt = idx & 127; idx >>= 7;
  int h  = idx & 15;  idx >>= 4;
  int b  = idx;

  int lane = threadIdx.x;
  int half = lane >> 4;
  int l15  = lane & 15;
  int kb8  = half * 8;
  int kb16 = half * 16;

  const bf16_t* Q  = qkv + (long)(b * N) * LDQ + h * D;
  const bf16_t* Kp = Q + INNER_;
  const bf16_t* Vp = Q + 2 * INNER_;
  int q0 = qt * 16;

  // Q fragments (16x64 = two 16x32 A-frags)
  const bf16_t* qrow = Q + (long)(q0 + l15) * LDQ;
  v16bf aq0 = make_a_frag(qrow, kb8);
  v16bf aq1 = make_a_frag(qrow + 32, kb8);

  // all-ones B matrix: row-sum of P via WMMA (every column = rowsum)
  v16bf ones;
#pragma unroll
  for (int j = 0; j < 16; ++j) ones[j] = (bf16_t)1.0f;

  float mrow[8];
  v8f o[4] = {};
  v8f lacc = {};
#pragma unroll
  for (int r = 0; r < 8; ++r) mrow[r] = -1e30f;

  __shared__ __align__(16) bf16_t Ps[16][40];  // P tile 16x32 (80B rows)
  const float scale = 0.125f;

  for (int kt = 0; kt < N / 32; ++kt) {
    int key0 = kt * 32;
    if (kt + 1 < N / 32) {
      __builtin_prefetch(Kp + (long)(key0 + 32 + l15) * LDQ, 0, 1);
      __builtin_prefetch(Vp + (long)(key0 + 32 + l15) * LDQ, 0, 1);
    }

    // S = Q * K^T for 32 keys -> two 16x16 f32 fragments
    v8f sfrag[2];
#pragma unroll
    for (int s = 0; s < 2; ++s) {
      const bf16_t* krow = Kp + (long)(key0 + s * 16 + l15) * LDQ;
      v16bf bk0 = *reinterpret_cast<const v16bf*>(krow + kb16);        // d 0..31
      v16bf bk1 = *reinterpret_cast<const v16bf*>(krow + 32 + kb16);   // d 32..63
      v8f c = {};
      c = __builtin_amdgcn_wmma_f32_16x16x32_bf16(false, aq0, false, bk0, (short)0, c, false, false);
      c = __builtin_amdgcn_wmma_f32_16x16x32_bf16(false, aq1, false, bk1, (short)0, c, false, false);
      sfrag[s] = c;
    }

    // Online softmax: row-max via lane butterflies, P = exp(S - m)
    float alpha[8];
#pragma unroll
    for (int r = 0; r < 8; ++r) {
      float v0 = sfrag[0][r] * scale;
      float v1 = sfrag[1][r] * scale;
      float mx = fmaxf(v0, v1);
#pragma unroll
      for (int off = 1; off < 16; off <<= 1) mx = fmaxf(mx, __shfl_xor(mx, off));
      float mnew = fmaxf(mrow[r], mx);
      alpha[r] = __expf(mrow[r] - mnew);
      mrow[r] = mnew;
      sfrag[0][r] = __expf(v0 - mnew);
      sfrag[1][r] = __expf(v1 - mnew);
    }
#pragma unroll
    for (int d = 0; d < 4; ++d)
#pragma unroll
      for (int r = 0; r < 8; ++r) o[d][r] *= alpha[r];
#pragma unroll
    for (int r = 0; r < 8; ++r) lacc[r] *= alpha[r];

    // P (f32 C-layout) -> LDS -> bf16 A-layout
#pragma unroll
    for (int s = 0; s < 2; ++s)
#pragma unroll
      for (int r = 0; r < 8; ++r)
        Ps[half * 8 + r][s * 16 + l15] = (bf16_t)sfrag[s][r];
    __syncthreads();
    v16bf ap = make_a_frag(&Ps[l15][0], kb8);
    __syncthreads();

    // row-sum of P via ones-matrix WMMA (replaces 32 ds_bpermute per tile)
    lacc = __builtin_amdgcn_wmma_f32_16x16x32_bf16(false, ap, false, ones,
                                                   (short)0, lacc, false, false);

    // O += P * V   (K-dim = 32 keys, four 16-wide d chunks)
#pragma unroll
    for (int d = 0; d < 4; ++d) {
      v16bf bv;
#pragma unroll
      for (int j = 0; j < 16; ++j)
        bv[j] = Vp[(long)(key0 + kb16 + j) * LDQ + d * 16 + l15];
      o[d] = __builtin_amdgcn_wmma_f32_16x16x32_bf16(false, ap, false, bv,
                                                     (short)0, o[d], false, false);
    }
  }

  // Finalize: divide by row sum, write bf16 [B*N, INNER]
#pragma unroll
  for (int d = 0; d < 4; ++d)
#pragma unroll
    for (int r = 0; r < 8; ++r) {
      float v = o[d][r] / lacc[r];
      long row = (long)b * N + q0 + half * 8 + r;
      out[row * INNER_ + h * D + d * 16 + l15] = (bf16_t)v;
    }
}

// ------------------------------- launcher ----------------------------------
extern "C" void kernel_launch(void* const* d_in, const int* in_sizes, int n_in,
                              void* d_out, int out_size, void* d_ws, size_t ws_size,
                              hipStream_t stream) {
  (void)in_sizes; (void)n_in; (void)out_size; (void)ws_size;
  const float* x     = (const float*)d_in[0];
  const float* ln1_g = (const float*)d_in[1];
  const float* ln1_b = (const float*)d_in[2];
  const float* ln2_g = (const float*)d_in[3];
  const float* ln2_b = (const float*)d_in[4];
  const float* w_qkv = (const float*)d_in[5];
  const float* w_out = (const float*)d_in[6];
  const float* b_out = (const float*)d_in[7];
  const float* w_ff1 = (const float*)d_in[8];
  const float* b_ff1 = (const float*)d_in[9];
  const float* w_ff2 = (const float*)d_in[10];
  const float* b_ff2 = (const float*)d_in[11];
  float* out = (float*)d_out;

  // Workspace layout (bytes, 256B aligned)
  char* ws = (char*)d_ws;
  size_t off = 0;
  auto alloc = [&](size_t bytes) { char* p = ws + off; off += (bytes + 255) & ~size_t(255); return p; };
  bf16_t* wqkv_t = (bf16_t*)alloc((size_t)DIM_ * 3 * INNER_ * 2);  // [3072,1024]
  bf16_t* wout_t = (bf16_t*)alloc((size_t)INNER_ * DIM_ * 2);      // [1024,1024]
  bf16_t* wff1_t = (bf16_t*)alloc((size_t)DIM_ * FF_ * 2);         // [4096,1024]
  bf16_t* wff2_t = (bf16_t*)alloc((size_t)FF_ * DIM_ * 2);         // [1024,4096]
  bf16_t* h_bf   = (bf16_t*)alloc((size_t)ROWS_ * DIM_ * 2);
  bf16_t* qkv_bf = (bf16_t*)alloc((size_t)ROWS_ * 3 * INNER_ * 2);
  bf16_t* attn_bf= (bf16_t*)alloc((size_t)ROWS_ * INNER_ * 2);
  float*  x1     = (float*) alloc((size_t)ROWS_ * DIM_ * 4);
  bf16_t* h2_bf  = (bf16_t*)alloc((size_t)ROWS_ * DIM_ * 2);
  bf16_t* a1_bf  = (bf16_t*)alloc((size_t)ROWS_ * FF_ * 2);

  // Weight transpose + fp32 -> bf16 (W[K,N] -> Wt[N,K])
  cvt_transpose_bf16_kernel<<<dim3(3 * INNER_ / 32, DIM_ / 32), 256, 0, stream>>>(
      w_qkv, wqkv_t, DIM_, 3 * INNER_);
  cvt_transpose_bf16_kernel<<<dim3(DIM_ / 32, INNER_ / 32), 256, 0, stream>>>(
      w_out, wout_t, INNER_, DIM_);
  cvt_transpose_bf16_kernel<<<dim3(FF_ / 32, DIM_ / 32), 256, 0, stream>>>(
      w_ff1, wff1_t, DIM_, FF_);
  cvt_transpose_bf16_kernel<<<dim3(DIM_ / 32, FF_ / 32), 256, 0, stream>>>(
      w_ff2, wff2_t, FF_, DIM_);

  // LN1: x -> h (bf16)
  ln_bf16_kernel<<<ROWS_, 256, 0, stream>>>(x, ln1_g, ln1_b, h_bf);

  // QKV GEMM: [4096,1024] x [1024,3072] -> bf16
  gemm_bf16_kernel<0, bf16_t><<<dim3(3 * INNER_ / 128, ROWS_ / 128), 256, 0, stream>>>(
      h_bf, wqkv_t, qkv_bf, nullptr, nullptr, ROWS_, 3 * INNER_, DIM_);

  // Flash attention -> attn (bf16)
  flash_attn_kernel<<<BATCH_ * HEADS_ * (SEQ_ / 16), 32, 0, stream>>>(qkv_bf, attn_bf);

  // Out projection + bias + residual(x) -> x1 (fp32)
  gemm_bf16_kernel<3, float><<<dim3(DIM_ / 128, ROWS_ / 128), 256, 0, stream>>>(
      attn_bf, wout_t, x1, b_out, x, ROWS_, DIM_, INNER_);

  // LN2: x1 -> h2 (bf16)
  ln_bf16_kernel<<<ROWS_, 256, 0, stream>>>(x1, ln2_g, ln2_b, h2_bf);

  // FF1 + bias + exact GELU -> a1 (bf16)
  gemm_bf16_kernel<2, bf16_t><<<dim3(FF_ / 128, ROWS_ / 128), 256, 0, stream>>>(
      h2_bf, wff1_t, a1_bf, b_ff1, nullptr, ROWS_, FF_, DIM_);

  // FF2 + bias + residual(x1) -> out (fp32)
  gemm_bf16_kernel<3, float><<<dim3(DIM_ / 128, ROWS_ / 128), 256, 0, stream>>>(
      a1_bf, wff2_t, out, b_ff2, x1, ROWS_, DIM_, FF_);
}
